// GNN_40819369181217
// MI455X (gfx1250) — matrix operand order repaired and checked by reading, not compiled, not run on hardware
//
#include <hip/hip_runtime.h>

typedef __attribute__((ext_vector_type(16))) _Float16 v16h;
typedef __attribute__((ext_vector_type(8)))  float    v8f;

#define B_ 64
#define S_ 512
#define D_ 256
#define L_ 3
#define SLOPE 0.01f
#define LDS_STRIDE 34   // 32 halves + 2 pad to spread LDS banks

// ---------------------------------------------------------------------------
// has_nb[b*S+i] = 1 if any adjacent[b][o][i] != 0 (sum over axis=1 > 0)
__global__ void k_hasnb(const float* __restrict__ adj, int* __restrict__ has_nb) {
    int idx = blockIdx.x * blockDim.x + threadIdx.x;   // over B*S
    int b = idx / S_, i = idx % S_;
    const float* a = adj + (size_t)b * S_ * S_ + i;
    int any = 0;
    for (int o = 0; o < S_; ++o) any |= (a[(size_t)o * S_] != 0.0f);
    has_nb[idx] = any;
}

// ---------------------------------------------------------------------------
// WT[l][n][k] = (f16) W[l][k][n]   (transposed so B-fragment k-pairs are contiguous)
__global__ void k_wcast(const float* __restrict__ W, _Float16* __restrict__ WT) {
    int idx = blockIdx.x * blockDim.x + threadIdx.x;   // over L*D*D
    int l = idx / (D_ * D_);
    int r = idx % (D_ * D_);
    int k = r / D_, n = r % D_;
    WT[(size_t)l * D_ * D_ + (size_t)n * D_ + k] = (_Float16)W[idx];
}

// xh = f16(x), elementwise (initial cast of nodes)
__global__ void k_cast(const float* __restrict__ x, _Float16* __restrict__ xh) {
    int idx = blockIdx.x * blockDim.x + threadIdx.x;
    xh[idx] = (_Float16)x[idx];
}

// ---------------------------------------------------------------------------
// xhT[b][d][s] = xh[b][s][d] : LDS-tiled transpose, coalesced on both sides.
// block (32,8), grid (D/32, S/32, B)
__global__ void k_transpose(const _Float16* __restrict__ src, _Float16* __restrict__ dst) {
    __shared__ _Float16 tile[32][34];
    int b  = blockIdx.z;
    int d0 = blockIdx.x * 32;
    int s0 = blockIdx.y * 32;
    int tx = threadIdx.x;   // 32
    int ty = threadIdx.y;   // 8
    const _Float16* sb = src + (size_t)b * S_ * D_;
    _Float16*       db = dst + (size_t)b * D_ * S_;
#pragma unroll
    for (int j = 0; j < 32; j += 8)
        tile[ty + j][tx] = sb[(size_t)(s0 + ty + j) * D_ + d0 + tx];
    __syncthreads();
#pragma unroll
    for (int j = 0; j < 32; j += 8)
        db[(size_t)(d0 + ty + j) * S_ + s0 + tx] = tile[tx][ty + j];
}

// ---------------------------------------------------------------------------
// pooled_h[b,i,d] = f16( sum_o (adj[b,o,i] != 0) * x[b,o,d] )
// Block = 128 threads (4 wave32), grid = B*(S/16). Wave w covers N cols [64w,64w+64).
// A tile (mask^T, 16x32) staged in LDS; B fragments from transposed xhT (k-pairs contiguous).
__global__ void k_pool(const float* __restrict__ adj, const _Float16* __restrict__ xhT,
                       _Float16* __restrict__ pooled_h) {
    __shared__ _Float16 Alds[16 * LDS_STRIDE];
    int tile = blockIdx.x;
    int b  = tile / (S_ / 16);
    int m0 = (tile % (S_ / 16)) * 16;
    int tid  = threadIdx.x;
    int wave = tid >> 5;
    int lane = tid & 31;
    int half = lane >> 4;
    int ml   = lane & 15;
    const float*    adjb = adj + (size_t)b * S_ * S_ + m0;   // column block m0
    const _Float16* xtb  = xhT + (size_t)b * D_ * S_;        // [d][o]
    v8f zero = {0.f,0.f,0.f,0.f,0.f,0.f,0.f,0.f};
    v8f acc[4];
    for (int t = 0; t < 4; ++t) acc[t] = zero;
    int n0w = wave * 64;

    for (int k0 = 0; k0 < S_; k0 += 32) {
        __syncthreads();   // previous-iteration fragment reads done
        // stage A(m, k=o) = (adj[b][k0+o][m0+m] != 0): 512 elems, 4 per thread
#pragma unroll
        for (int j = 0; j < 4; ++j) {
            int e  = tid + 128 * j;
            int mm = e & 15;     // consecutive tids -> consecutive columns: coalesced
            int oo = e >> 4;
            float v = adjb[(size_t)(k0 + oo) * S_ + mm];
            Alds[mm * LDS_STRIDE + oo] = (_Float16)(v != 0.0f ? 1.0f : 0.0f);
        }
        __syncthreads();
        v16h a;   // A fragment from LDS (ds_load_b32 pairs)
#pragma unroll
        for (int v = 0; v < 8; ++v) {
            int kb = (v < 4) ? (2*v + 8*half) : (8 + 2*v + 8*half);
            a[2*v]     = Alds[ml * LDS_STRIDE + kb];
            a[2*v + 1] = Alds[ml * LDS_STRIDE + kb + 1];
        }
#pragma unroll
        for (int t = 0; t < 4; ++t) {
            int n = n0w + t * 16 + ml;
            v16h bb;   // B(k=o, n=d) = xhT[d][o]; pairs contiguous -> b32 loads
#pragma unroll
            for (int v = 0; v < 8; ++v) {
                int kb = (v < 4) ? (2*v + 8*half) : (8 + 2*v + 8*half);
                const _Float16* pr = xtb + (size_t)n * S_ + k0 + kb;
                bb[2*v]     = pr[0];
                bb[2*v + 1] = pr[1];
            }
            acc[t] = __builtin_amdgcn_wmma_f32_16x16x32_f16(
                false, a, false, bb, (short)0, acc[t], false, false);
        }
    }
    _Float16* pb = pooled_h + (size_t)b * S_ * D_;
#pragma unroll
    for (int t = 0; t < 4; ++t) {
        int n = n0w + t * 16 + ml;
#pragma unroll
        for (int r = 0; r < 8; ++r) {
            int m = m0 + r + 8 * half;
            pb[(size_t)m * D_ + n] = (_Float16)acc[t][r];
        }
    }
}

// ---------------------------------------------------------------------------
// Dual FFN + select:
//   val = has_nb[b,m] ? lrelu(pooled@Wp + bp) : lrelu(x@Wn + bn)
//   xh_out = f16(val) always; xf32 (optional) = val
__global__ void k_ffn(const _Float16* __restrict__ xh, const _Float16* __restrict__ ph,
                      const _Float16* __restrict__ WnT, const _Float16* __restrict__ WpT,
                      const float* __restrict__ bn, const float* __restrict__ bp,
                      const int* __restrict__ has_nb,
                      _Float16* __restrict__ xh_out, float* __restrict__ xf32) {
    __shared__ _Float16 ANlds[16 * LDS_STRIDE];
    __shared__ _Float16 APlds[16 * LDS_STRIDE];
    int tile = blockIdx.x;
    int b  = tile / (S_ / 16);
    int m0 = (tile % (S_ / 16)) * 16;
    int tid  = threadIdx.x;
    int wave = tid >> 5;
    int lane = tid & 31;
    int half = lane >> 4;
    int ml   = lane & 15;
    const _Float16* xb = xh + (size_t)b * S_ * D_;
    const _Float16* pb = ph + (size_t)b * S_ * D_;
    v8f zero = {0.f,0.f,0.f,0.f,0.f,0.f,0.f,0.f};
    v8f accN[4], accP[4];
    for (int t = 0; t < 4; ++t) { accN[t] = zero; accP[t] = zero; }
    int n0w = wave * 64;

    for (int k0 = 0; k0 < D_; k0 += 32) {
        __syncthreads();
        // stage both A tiles [16x32] from f16 activations: coalesced along k
#pragma unroll
        for (int j = 0; j < 4; ++j) {
            int e  = tid + 128 * j;
            int kk = e & 31;
            int mm = e >> 5;
            ANlds[mm * LDS_STRIDE + kk] = xb[(size_t)(m0 + mm) * D_ + k0 + kk];
            APlds[mm * LDS_STRIDE + kk] = pb[(size_t)(m0 + mm) * D_ + k0 + kk];
        }
        __syncthreads();
        v16h aN, aP;
#pragma unroll
        for (int v = 0; v < 8; ++v) {
            int kb = (v < 4) ? (2*v + 8*half) : (8 + 2*v + 8*half);
            aN[2*v]     = ANlds[ml * LDS_STRIDE + kb];
            aN[2*v + 1] = ANlds[ml * LDS_STRIDE + kb + 1];
            aP[2*v]     = APlds[ml * LDS_STRIDE + kb];
            aP[2*v + 1] = APlds[ml * LDS_STRIDE + kb + 1];
        }
#pragma unroll
        for (int t = 0; t < 4; ++t) {
            int n = n0w + t * 16 + ml;
            v16h bN, bP;   // B fragments from transposed f16 weights: WT[n][k]
#pragma unroll
            for (int v = 0; v < 8; ++v) {
                int kb = (v < 4) ? (2*v + 8*half) : (8 + 2*v + 8*half);
                const _Float16* wr = WnT + (size_t)n * D_ + k0 + kb;
                bN[2*v] = wr[0]; bN[2*v + 1] = wr[1];
                const _Float16* wq = WpT + (size_t)n * D_ + k0 + kb;
                bP[2*v] = wq[0]; bP[2*v + 1] = wq[1];
            }
            accN[t] = __builtin_amdgcn_wmma_f32_16x16x32_f16(
                false, aN, false, bN, (short)0, accN[t], false, false);
            accP[t] = __builtin_amdgcn_wmma_f32_16x16x32_f16(
                false, aP, false, bP, (short)0, accP[t], false, false);
        }
    }
    _Float16* oh = xh_out + (size_t)b * S_ * D_;
    const int* hb = has_nb + b * S_;
#pragma unroll
    for (int t = 0; t < 4; ++t) {
        int n = n0w + t * 16 + ml;
        float biasN = bn[n], biasP = bp[n];
#pragma unroll
        for (int r = 0; r < 8; ++r) {
            int m = m0 + r + 8 * half;
            float vn = accN[t][r] + biasN; vn = (vn >= 0.f) ? vn : SLOPE * vn;
            float vp = accP[t][r] + biasP; vp = (vp >= 0.f) ? vp : SLOPE * vp;
            float val = hb[m] ? vp : vn;
            oh[(size_t)m * D_ + n] = (_Float16)val;
            if (xf32) xf32[(size_t)b * S_ * D_ + (size_t)m * D_ + n] = val;
        }
    }
}

// ---------------------------------------------------------------------------
// g[b,d] = sum_s x[b,s,d]
__global__ void k_gpool(const float* __restrict__ x, float* __restrict__ g) {
    int b = blockIdx.x;
    int d = threadIdx.x;   // 256
    const float* xb = x + (size_t)b * S_ * D_ + d;
    float s = 0.f;
    for (int o = 0; o < S_; ++o) s += xb[(size_t)o * D_];
    g[b * D_ + d] = s;
}

// out[b,n] = lrelu(sum_k g[b,k]*W[k,n] + bias[n])   (tiny: 8.4 MFLOP, fp32)
__global__ void k_gffn(const float* __restrict__ g, const float* __restrict__ W,
                       const float* __restrict__ bias, float* __restrict__ out) {
    int b = blockIdx.x;
    int n = threadIdx.x;   // 256
    const float* gb = g + b * D_;
    float s = bias[n];
    for (int k = 0; k < D_; ++k) s += gb[k] * W[(size_t)k * D_ + n];
    out[b * D_ + n] = (s >= 0.f) ? s : SLOPE * s;
}

// ---------------------------------------------------------------------------
extern "C" void kernel_launch(void* const* d_in, const int* in_sizes, int n_in,
                              void* d_out, int out_size, void* d_ws, size_t ws_size,
                              hipStream_t stream) {
    (void)in_sizes; (void)n_in; (void)out_size; (void)ws_size;
    const float* nodes   = (const float*)d_in[0];
    const float* adj     = (const float*)d_in[1];
    const float* W_node  = (const float*)d_in[2];
    const float* b_node  = (const float*)d_in[3];
    const float* W_neigh = (const float*)d_in[4];
    const float* b_neigh = (const float*)d_in[5];
    const float* W_glob  = (const float*)d_in[6];
    const float* b_glob  = (const float*)d_in[7];

    char* ws = (char*)d_ws;
    const size_t NXH = (size_t)B_ * S_ * D_ * sizeof(_Float16);  // 16 MiB
    // workspace layout (bytes)
    int*      has_nb   = (int*)     (ws + 0);                            // 131072
    _Float16* WnT      = (_Float16*)(ws + 131072);                       // 393216
    _Float16* WpT      = (_Float16*)(ws + 524288);                       // 393216
    float*    g        = (float*)   (ws + 917504);                       // 65536
    _Float16* xh0      = (_Float16*)(ws + 983040);                       // 16 MiB
    _Float16* xh1      = (_Float16*)(ws + 983040 + NXH);                 // 16 MiB
    _Float16* pooled_h = (_Float16*)(ws + 983040 + 2 * NXH);             // 16 MiB
    _Float16* xhT      = (_Float16*)(ws + 983040 + 3 * NXH);             // 16 MiB
    float*    x2       = (float*)   (ws + 983040 + 4 * NXH);             // 32 MiB

    float* xout = (float*)d_out;                         // [B,S,D]
    float* gout = (float*)d_out + (size_t)B_ * S_ * D_;  // [B,D]

    k_hasnb<<<(B_ * S_) / 256, 256, 0, stream>>>(adj, has_nb);
    k_wcast<<<(L_ * D_ * D_) / 256, 256, 0, stream>>>(W_node, WnT);
    k_wcast<<<(L_ * D_ * D_) / 256, 256, 0, stream>>>(W_neigh, WpT);
    k_cast <<<(B_ * S_ * D_) / 256, 256, 0, stream>>>(nodes, xh0);

    dim3 gtile(B_ * (S_ / 16));
    dim3 tgrid(D_ / 32, S_ / 32, B_);
    dim3 tblk(32, 8);
    _Float16* xh_io[4]  = {xh0, xh1, xh0, xh1};     // layer l: in xh_io[l], out xh_io[l+1]
    float*    xf32_o[3] = {nullptr, x2, xout};      // fp32 copy only where needed
    for (int l = 0; l < L_; ++l) {
        k_transpose<<<tgrid, tblk, 0, stream>>>(xh_io[l], xhT);
        k_pool<<<gtile, 128, 0, stream>>>(adj, xhT, pooled_h);
        k_ffn <<<gtile, 128, 0, stream>>>(xh_io[l], pooled_h,
                                          WnT + (size_t)l * D_ * D_,
                                          WpT + (size_t)l * D_ * D_,
                                          b_node + (size_t)l * D_,
                                          b_neigh + (size_t)l * D_,
                                          has_nb, xh_io[l + 1], xf32_o[l]);
    }
    // global branch uses x entering the last layer (= output of layer 1)
    k_gpool<<<B_, D_, 0, stream>>>(x2, g);
    k_gffn <<<B_, D_, 0, stream>>>(g, W_glob, b_glob, gout);
}
